// SetTransformer_9208409883007
// MI455X (gfx1250) — matrix operand order repaired and checked by reading, not compiled, not run on hardware
//
#include <hip/hip_runtime.h>
#include <hip/hip_bf16.h>
#include <math.h>

// ---------------------------------------------------------------------------
// SetTransformer (linear-attention encoder x4 + seed attention pooling)
// bf16-resident operands: weights converted once, activations written as bf16
// by the GEMM epilogue; x kept fp32+bf16 for the residual/LN chain.
// GEMM core: v_wmma_f32_16x16x32_bf16, LDS double-buffered, staging is pure
// 16B copies (async global->LDS on toolchains that expose the builtin).
// ---------------------------------------------------------------------------

typedef __bf16 v16bf __attribute__((ext_vector_type(16)));
typedef float  v8f   __attribute__((ext_vector_type(8)));

#define BM 128
#define BN 128
#define BKD 32
#define LDA 40    // BKD + 8 pad : 80B rows (16B aligned, 20-dword bank rotate)
#define LDB 136   // BN  + 8 pad : 272B rows (16B aligned, 68-dword rotate)
#define NTHREADS 256

#if defined(__has_builtin)
# if __has_builtin(__builtin_amdgcn_global_load_async_to_lds_b128)
#  define ASYNC_LDS 1
# endif
#endif
#ifndef ASYNC_LDS
# define ASYNC_LDS 0
#endif

#if ASYNC_LDS
typedef int v4i_ldst __attribute__((vector_size(16)));
typedef __attribute__((address_space(1))) v4i_ldst* g_v4i_p;
typedef __attribute__((address_space(3))) v4i_ldst* l_v4i_p;
__device__ __forceinline__ void async_cp16(const unsigned short* gp,
                                           unsigned short* lp) {
  __builtin_amdgcn_global_load_async_to_lds_b128((g_v4i_p)gp, (l_v4i_p)lp, 0, 0);
}
#endif

struct GemmP {
  const unsigned short* A; long a_rs, a_cs, a_bs0, a_bs1;  // bf16 A(m,k) strided
  const unsigned short* W; long w_rs, w_bs0, w_bs1;        // bf16 W(k,n), n contig
  const float* bias;                                       // [N] or null
  const float* R;  long r_rs, r_bs0, r_bs1;                // fp32 residual or null
  const float* rs; long rs_st, rs_bs0, rs_bs1;             // per-row scale or null
  void* C; long c_rs, c_bs0, c_bs1;                        // fp32 or bf16 out
  int M, N, K, act, inner, c_bf16;                         // act: 0 none,1 relu,2 elu+1
};

__device__ __forceinline__ unsigned int f2bf_bits(float f) {
  unsigned int u = __float_as_uint(f);
  u += 0x7FFFu + ((u >> 16) & 1u);                 // round-to-nearest-even
  return u >> 16;
}
__device__ __forceinline__ unsigned short f2bf(float f) {
  return (unsigned short)f2bf_bits(f);
}
__device__ __forceinline__ float bf2f(unsigned short h) {
  return __uint_as_float(((unsigned int)h) << 16);
}

__device__ __forceinline__ v16bf frag16(const unsigned short* p0,
                                        const unsigned short* p1) {
  union { uint4 u[2]; v16bf f; } c;
  c.u[0] = *(const uint4*)p0;
  c.u[1] = *(const uint4*)p1;
  return c.f;
}

__device__ __forceinline__ void tile_sync() {
#if ASYNC_LDS
# if defined(__has_builtin) && __has_builtin(__builtin_amdgcn_s_wait_asynccnt)
  __builtin_amdgcn_s_wait_asynccnt(0);
# else
  asm volatile("s_wait_asynccnt 0x0" ::: "memory");
# endif
#endif
  __syncthreads();
}

__global__ __launch_bounds__(NTHREADS)
void gemm_bf16_wmma(GemmP p) {
  __shared__ __align__(16) unsigned short sA[2][BM * LDA];
  __shared__ __align__(16) unsigned short sB[2][BKD * LDB];

  const int tid  = threadIdx.x;
  const int lane = tid & 31;
  const int wave = tid >> 5;
  const int wm   = wave >> 1;     // 4 row-groups of 32
  const int wn   = wave & 1;      // 2 col-groups of 64

  const int z  = blockIdx.z;
  const int bz = z / p.inner;
  const int hz = z - bz * p.inner;

  const unsigned short* Ab = p.A + bz * p.a_bs0 + hz * p.a_bs1;
  const unsigned short* Wb = p.W + bz * p.w_bs0 + hz * p.w_bs1;

  const int m0 = blockIdx.y * BM;
  const int n0 = blockIdx.x * BN;

  const bool fullK = (p.K & (BKD - 1)) == 0;
  const bool fastA = fullK && (p.a_cs == 1) && (m0 + BM <= p.M);
  const bool fastB = fullK && (n0 + BN <= p.N);

  // ---- staging: full tiles are pure 16B/lane copies (async if available) ---
  auto stageA = [&](unsigned short* dst, int k0) {
    if (fastA) {
#pragma unroll
      for (int pass = 0; pass < 2; ++pass) {
        const int chunk = tid + pass * 256;          // 512 x 16B chunks
        const int r = chunk >> 2, c = (chunk & 3) * 8;
        const unsigned short* gp = Ab + (long)(m0 + r) * p.a_rs + (k0 + c);
#if ASYNC_LDS
        async_cp16(gp, &dst[r * LDA + c]);
#else
        *(uint4*)&dst[r * LDA + c] = *(const uint4*)gp;
#endif
      }
    } else {
#pragma unroll
      for (int i = tid; i < BM * BKD; i += NTHREADS) {
        int r = i >> 5, c = i & 31;
        int gm = m0 + r, gk = k0 + c;
        unsigned short v = 0;
        if (gm < p.M && gk < p.K) v = Ab[(long)gm * p.a_rs + (long)gk * p.a_cs];
        dst[r * LDA + c] = v;
      }
    }
  };
  auto stageB = [&](unsigned short* dst, int k0) {
    if (fastB) {
#pragma unroll
      for (int pass = 0; pass < 2; ++pass) {
        const int chunk = tid + pass * 256;
        const int r = chunk >> 4, c = (chunk & 15) * 8;
        const unsigned short* gp = Wb + (long)(k0 + r) * p.w_rs + (n0 + c);
#if ASYNC_LDS
        async_cp16(gp, &dst[r * LDB + c]);
#else
        *(uint4*)&dst[r * LDB + c] = *(const uint4*)gp;
#endif
      }
    } else {
#pragma unroll
      for (int i = tid; i < BKD * BN; i += NTHREADS) {
        int r = i >> 7, c = i & 127;
        int gk = k0 + r, gn = n0 + c;
        unsigned short v = 0;
        if (gk < p.K && gn < p.N) v = Wb[(long)gk * p.w_rs + gn];
        dst[r * LDB + c] = v;
      }
    }
  };

  v8f zero = {0.f,0.f,0.f,0.f,0.f,0.f,0.f,0.f};
  v8f acc[2][4];
#pragma unroll
  for (int t = 0; t < 2; ++t)
#pragma unroll
    for (int u = 0; u < 4; ++u) acc[t][u] = zero;

  const int ksteps = (p.K + BKD - 1) / BKD;

  stageA(sA[0], 0);
  stageB(sB[0], 0);
  tile_sync();

  for (int ks = 0; ks < ksteps; ++ks) {
    const int cur = ks & 1;
    if (ks + 1 < ksteps) {            // prefetch next tile into the other buffer
      stageA(sA[cur ^ 1], (ks + 1) * BKD);
      stageB(sB[cur ^ 1], (ks + 1) * BKD);
    }

    // ---- gather fragments per ISA 7.12.2 layouts ----
    v16bf afr[2], bfr[4];
    const int arow = wm * 32 + (lane & 15);
    const int kb   = (lane >> 4) << 3;           // 0 or 8
#pragma unroll
    for (int t = 0; t < 2; ++t) {
      const unsigned short* pa = &sA[cur][(arow + t * 16) * LDA + kb];
      afr[t] = frag16(pa, pa + 16);              // K {kb..kb+7, kb+16..kb+23}
    }
#pragma unroll
    for (int u = 0; u < 4; ++u) {
      const unsigned short* pb = &sB[cur][wn * 64 + u * 16 + (lane & 15)];
      union { unsigned short h[16]; v16bf f; } cb;
#pragma unroll
      for (int s = 0; s < 16; ++s)
        cb.h[s] = pb[(kb + (s & 7) + ((s >> 3) << 4)) * LDB];
      bfr[u] = cb.f;                             // N=lane&15, K {kb..kb+7, kb+16..kb+23}
    }
#pragma unroll
    for (int t = 0; t < 2; ++t)
#pragma unroll
      for (int u = 0; u < 4; ++u)
        acc[t][u] = __builtin_amdgcn_wmma_f32_16x16x32_bf16(
            false, afr[t], false, bfr[u], (short)0, acc[t][u], false, false);
    tile_sync();
  }

  // ---- fused epilogue ----
  const float* Rb = p.R  ? p.R  + bz * p.r_bs0  + hz * p.r_bs1  : nullptr;
  const float* Sb = p.rs ? p.rs + bz * p.rs_bs0 + hz * p.rs_bs1 : nullptr;
  float*          Cf = p.c_bf16 ? nullptr
                                : (float*)p.C + bz * p.c_bs0 + hz * p.c_bs1;
  unsigned short* Ch = p.c_bf16 ? (unsigned short*)p.C + bz * p.c_bs0 + hz * p.c_bs1
                                : nullptr;

#pragma unroll
  for (int t = 0; t < 2; ++t)
#pragma unroll
    for (int u = 0; u < 4; ++u)
#pragma unroll
      for (int r = 0; r < 8; ++r) {
        int ml = wm * 32 + t * 16 + r + ((lane >> 4) << 3);
        int nl = wn * 64 + u * 16 + (lane & 15);
        int gm = m0 + ml, gn = n0 + nl;
        if (gm < p.M && gn < p.N) {
          float v = acc[t][u][r];
          if (p.bias) v += p.bias[gn];
          if (Sb)     v *= Sb[(long)gm * p.rs_st];
          if (p.act == 1)      v = fmaxf(v, 0.f);
          else if (p.act == 2) v = (v > 0.f) ? (v + 1.f) : __expf(v);
          if (Rb) v += Rb[(long)gm * p.r_rs + gn];
          long idx = (long)gm * p.c_rs + gn;
          if (Ch) Ch[idx] = f2bf(v); else Cf[idx] = v;
        }
      }
}

// ---- fp32 -> bf16 bulk convert ----
__global__ __launch_bounds__(256)
void cvt_f32_bf16(const float* src, unsigned short* dst, long n) {
  long i = ((long)blockIdx.x * 256 + threadIdx.x) * 4;
  if (i + 3 < n) {
    float4 f = *(const float4*)(src + i);
    uint2 dd;
    dd.x = f2bf_bits(f.x) | (f2bf_bits(f.y) << 16);
    dd.y = f2bf_bits(f.z) | (f2bf_bits(f.w) << 16);
    *(uint2*)(dst + i) = dd;
  } else {
    for (long j = i; j < n; ++j) dst[j] = f2bf(src[j]);
  }
}

// ---- LayerNorm over D=512, one block per row; writes fp32 x and bf16 xh ----
__global__ __launch_bounds__(256)
void ln_kernel(float* x, unsigned short* xh, const float* g, const float* b) {
  long row = blockIdx.x;
  float* xr = x + row * 512;
  unsigned short* hr = xh + row * 512;
  int t = threadIdx.x;
  float v0 = xr[t], v1 = xr[t + 256];
  __shared__ float red[256];
  red[t] = v0 + v1;
  __syncthreads();
  for (int o = 128; o > 0; o >>= 1) { if (t < o) red[t] += red[t + o]; __syncthreads(); }
  float mean = red[0] * (1.f / 512.f);
  __syncthreads();
  float d0 = v0 - mean, d1 = v1 - mean;
  red[t] = d0 * d0 + d1 * d1;
  __syncthreads();
  for (int o = 128; o > 0; o >>= 1) { if (t < o) red[t] += red[t + o]; __syncthreads(); }
  float inv = rsqrtf(red[0] * (1.f / 512.f) + 1e-5f);
  float o0 = d0 * inv * g[t] + b[t];
  float o1 = d1 * inv * g[t + 256] + b[t + 256];
  xr[t] = o0;        xr[t + 256] = o1;
  hr[t] = f2bf(o0);  hr[t + 256] = f2bf(o1);
}

// ---- ksum[b, d] = sum_s kf[b, s, d] (kf bf16) : grid (D/256, B, 8) ----
__global__ __launch_bounds__(256)
void colsum_kernel(const unsigned short* kf, float* ksum) {
  int d  = blockIdx.x * 256 + threadIdx.x;
  int b  = blockIdx.y;
  int s0 = blockIdx.z * 512;
  const unsigned short* p = kf + ((long)b * 4096 + s0) * 512 + d;
  float a = 0.f;
  for (int s = 0; s < 512; ++s) a += bf2f(p[(long)s * 512]);
  atomicAdd(&ksum[b * 512 + d], a);
}

// ---- z[b,r,h] = 1/(sum_d qf[b?,r,h*64+d] * ksum[b,h*64+d] + eps) ----
__global__ __launch_bounds__(256)
void zden_kernel(const unsigned short* q, const float* ksum, float* z,
                 int rows, long qbs, int total) {
  int idx = blockIdx.x * blockDim.x + threadIdx.x;
  if (idx >= total) return;
  int h = idx & 7;
  int r = (idx >> 3) % rows;
  int b = idx / (rows * 8);
  const unsigned short* qp = q + (long)b * qbs + (long)r * 512 + h * 64;
  const float* kp = ksum + (long)b * 512 + h * 64;
  float s = 0.f;
#pragma unroll 8
  for (int d = 0; d < 64; ++d) s += bf2f(qp[d]) * kp[d];
  z[idx] = 1.f / (s + 1e-6f);
}

// ---------------------------------------------------------------------------
// Host orchestration
// ---------------------------------------------------------------------------
static const long Bc = 8, Sc = 4096, Dc = 512, Hc = 8, NL = 4, Kc = 32, FFc = 2048;
static const long Rrows = Bc * Sc;               // 32768

static void cvt(hipStream_t st, const float* s, unsigned short* d, long n) {
  long th = (n + 3) / 4;
  cvt_f32_bf16<<<(unsigned)((th + 255) / 256), 256, 0, st>>>(s, d, n);
}

static void gemm_plain(hipStream_t st, const unsigned short* A, long a_rs,
                       const unsigned short* W, long w_rs, const float* bias,
                       const float* R, long r_rs,
                       void* C, int c_bf16, long c_rs, int M, int N, int K, int act) {
  GemmP p{};
  p.A = A; p.a_rs = a_rs; p.a_cs = 1;
  p.W = W; p.w_rs = w_rs;
  p.bias = bias;
  p.R = R; p.r_rs = r_rs;
  p.C = C; p.c_rs = c_rs; p.c_bf16 = c_bf16;
  p.M = M; p.N = N; p.K = K; p.act = act; p.inner = 1;
  dim3 g((unsigned)((N + BN - 1) / BN), (unsigned)((M + BM - 1) / BM), 1);
  gemm_bf16_wmma<<<g, NTHREADS, 0, st>>>(p);
}

extern "C" void kernel_launch(void* const* d_in, const int* in_sizes, int n_in,
                              void* d_out, int out_size, void* d_ws, size_t ws_size,
                              hipStream_t stream) {
  (void)in_sizes; (void)n_in; (void)out_size; (void)ws_size;
  const float* src   = (const float*)d_in[0];
  const float* seeds = (const float*)d_in[1];
  const float* Wq = (const float*)d_in[2];  const float* bq = (const float*)d_in[3];
  const float* Wk = (const float*)d_in[4];  const float* bk = (const float*)d_in[5];
  const float* Wv = (const float*)d_in[6];  const float* bv = (const float*)d_in[7];
  const float* Wo = (const float*)d_in[8];  const float* bo = (const float*)d_in[9];
  const float* ln1g = (const float*)d_in[10]; const float* ln1b = (const float*)d_in[11];
  const float* ln2g = (const float*)d_in[12]; const float* ln2b = (const float*)d_in[13];
  const float* W1 = (const float*)d_in[14]; const float* b1 = (const float*)d_in[15];
  const float* W2 = (const float*)d_in[16]; const float* b2 = (const float*)d_in[17];
  const float* pWq = (const float*)d_in[18]; const float* pbq = (const float*)d_in[19];
  const float* pWk = (const float*)d_in[20]; const float* pbk = (const float*)d_in[21];
  const float* pWv = (const float*)d_in[22]; const float* pbv = (const float*)d_in[23];
  const float* pWo = (const float*)d_in[24]; const float* pbo = (const float*)d_in[25];

  const long TD = Rrows * Dc;                    // 16,777,216 elements
  char* wsb = (char*)d_ws;
  // fp32 region
  float* xb  = (float*)wsb;                 wsb += TD * 4;            // x fp32
  float* ksb = (float*)wsb;                 wsb += Bc * Dc * 4;       // ksum
  float* zb  = (float*)wsb;                 wsb += Bc * Sc * Hc * 4;  // z
  // bf16 region
  unsigned short* xh  = (unsigned short*)wsb; wsb += TD * 2;          // x bf16
  unsigned short* qh  = (unsigned short*)wsb; wsb += TD * 2;          // qf
  unsigned short* kh  = (unsigned short*)wsb; wsb += TD * 2;          // kf
  unsigned short* vh  = (unsigned short*)wsb; wsb += TD * 2;          // v
  unsigned short* oh  = (unsigned short*)wsb; wsb += TD * 2;          // o / ffn hidden
  unsigned short* kvh = (unsigned short*)wsb; wsb += Bc * Hc * 64 * 64 * 2;
  unsigned short* sh  = (unsigned short*)wsb; wsb += Kc * Dc * 2;     // qf(seeds)
  unsigned short* seedh = (unsigned short*)wsb; wsb += Kc * Dc * 2;   // seeds bf16
  unsigned short* wqh = (unsigned short*)wsb; wsb += NL * Dc * Dc * 2;
  unsigned short* wkh = (unsigned short*)wsb; wsb += NL * Dc * Dc * 2;
  unsigned short* wvh = (unsigned short*)wsb; wsb += NL * Dc * Dc * 2;
  unsigned short* woh = (unsigned short*)wsb; wsb += NL * Dc * Dc * 2;
  unsigned short* w1h = (unsigned short*)wsb; wsb += NL * Dc * FFc * 2;
  unsigned short* w2h = (unsigned short*)wsb; wsb += NL * FFc * Dc * 2;
  unsigned short* pwqh = (unsigned short*)wsb; wsb += Dc * Dc * 2;
  unsigned short* pwkh = (unsigned short*)wsb; wsb += Dc * Dc * 2;
  unsigned short* pwvh = (unsigned short*)wsb; wsb += Dc * Dc * 2;
  unsigned short* pwoh = (unsigned short*)wsb; wsb += Dc * Dc * 2;

  // one-time conversions (weights, seeds, x)
  cvt(stream, Wq, wqh, NL * Dc * Dc);
  cvt(stream, Wk, wkh, NL * Dc * Dc);
  cvt(stream, Wv, wvh, NL * Dc * Dc);
  cvt(stream, Wo, woh, NL * Dc * Dc);
  cvt(stream, W1, w1h, NL * Dc * FFc);
  cvt(stream, W2, w2h, NL * FFc * Dc);
  cvt(stream, pWq, pwqh, Dc * Dc);
  cvt(stream, pWk, pwkh, Dc * Dc);
  cvt(stream, pWv, pwvh, Dc * Dc);
  cvt(stream, pWo, pwoh, Dc * Dc);
  cvt(stream, seeds, seedh, Kc * Dc);
  cvt(stream, src, xh, TD);
  (void)hipMemcpyAsync(xb, src, (size_t)TD * sizeof(float),
                       hipMemcpyDeviceToDevice, stream);

  for (int L = 0; L < NL; ++L) {
    const unsigned short* wq = wqh + (long)L * Dc * Dc; const float* bq_ = bq + (long)L * Dc;
    const unsigned short* wk = wkh + (long)L * Dc * Dc; const float* bk_ = bk + (long)L * Dc;
    const unsigned short* wv = wvh + (long)L * Dc * Dc; const float* bv_ = bv + (long)L * Dc;
    const unsigned short* wo = woh + (long)L * Dc * Dc; const float* bo_ = bo + (long)L * Dc;
    const unsigned short* w1 = w1h + (long)L * Dc * FFc; const float* b1_ = b1 + (long)L * FFc;
    const unsigned short* w2 = w2h + (long)L * FFc * Dc; const float* b2_ = b2 + (long)L * Dc;

    // qf = elu(x@Wq+bq)+1 ; kf likewise ; v plain (all bf16 outputs)
    gemm_plain(stream, xh, Dc, wq, Dc, bq_, nullptr, 0, qh, 1, Dc, Rrows, Dc, Dc, 2);
    gemm_plain(stream, xh, Dc, wk, Dc, bk_, nullptr, 0, kh, 1, Dc, Rrows, Dc, Dc, 2);
    gemm_plain(stream, xh, Dc, wv, Dc, bv_, nullptr, 0, vh, 1, Dc, Rrows, Dc, Dc, 0);

    // ksum
    (void)hipMemsetAsync(ksb, 0, Bc * Dc * sizeof(float), stream);
    colsum_kernel<<<dim3(2, 8, 8), 256, 0, stream>>>(kh, ksb);

    // kv[b,h] = kf^T @ v  (64x64, K=S) batched over (b,h) -> bf16
    {
      GemmP p{};
      p.A = kh; p.a_rs = 1; p.a_cs = Dc; p.a_bs0 = Sc * Dc; p.a_bs1 = 64;
      p.W = vh; p.w_rs = Dc; p.w_bs0 = Sc * Dc; p.w_bs1 = 64;
      p.C = kvh; p.c_rs = 64; p.c_bs0 = Hc * 64 * 64; p.c_bs1 = 64 * 64; p.c_bf16 = 1;
      p.M = 64; p.N = 64; p.K = (int)Sc; p.act = 0; p.inner = (int)Hc;
      gemm_bf16_wmma<<<dim3(1, 1, (unsigned)(Bc * Hc)), NTHREADS, 0, stream>>>(p);
    }

    // z = 1/(qf . ksum + eps)
    {
      int total = (int)(Bc * Sc * Hc);
      zden_kernel<<<(total + 255) / 256, 256, 0, stream>>>(
          qh, ksb, zb, (int)Sc, Sc * Dc, total);
    }

    // o[b,s,h,:] = z * (qf @ kv)  batched over (b,h) -> bf16
    {
      GemmP p{};
      p.A = qh; p.a_rs = Dc; p.a_cs = 1; p.a_bs0 = Sc * Dc; p.a_bs1 = 64;
      p.W = kvh; p.w_rs = 64; p.w_bs0 = Hc * 64 * 64; p.w_bs1 = 64 * 64;
      p.rs = zb; p.rs_st = Hc; p.rs_bs0 = Sc * Hc; p.rs_bs1 = 1;
      p.C = oh; p.c_rs = Dc; p.c_bs0 = Sc * Dc; p.c_bs1 = 64; p.c_bf16 = 1;
      p.M = (int)Sc; p.N = 64; p.K = 64; p.act = 0; p.inner = (int)Hc;
      gemm_bf16_wmma<<<dim3(1, 32, (unsigned)(Bc * Hc)), NTHREADS, 0, stream>>>(p);
    }

    // x = x + o@Wo + bo  (fp32) ;  x = LN1(x) (dual-write)
    gemm_plain(stream, oh, Dc, wo, Dc, bo_, xb, Dc, xb, 0, Dc, Rrows, Dc, Dc, 0);
    ln_kernel<<<(unsigned)Rrows, 256, 0, stream>>>(xb, xh,
        ln1g + (long)L * Dc, ln1b + (long)L * Dc);

    // FFN in 4 row-chunks (bf16 hidden reuses oh): x = LN2(x + relu(x@W1+b1)@W2+b2)
    for (int c = 0; c < 4; ++c) {
      float* xc = xb + (long)c * 8192 * Dc;
      unsigned short* xhc = xh + (long)c * 8192 * Dc;
      gemm_plain(stream, xhc, Dc, w1, FFc, b1_, nullptr, 0, oh, 1, FFc, 8192, (int)FFc, (int)Dc, 1);
      gemm_plain(stream, oh, FFc, w2, Dc, b2_, xc, Dc, xc, 0, Dc, 8192, (int)Dc, (int)FFc, 0);
    }
    ln_kernel<<<(unsigned)Rrows, 256, 0, stream>>>(xb, xh,
        ln2g + (long)L * Dc, ln2b + (long)L * Dc);
  }

  // ---------------- seed attention pooling ----------------
  gemm_plain(stream, seedh, Dc, pwqh, Dc, pbq, nullptr, 0, sh, 1, Dc, (int)Kc, (int)Dc, (int)Dc, 2);
  gemm_plain(stream, xh, Dc, pwkh, Dc, pbk, nullptr, 0, kh, 1, Dc, Rrows, Dc, Dc, 2);
  gemm_plain(stream, xh, Dc, pwvh, Dc, pbv, nullptr, 0, vh, 1, Dc, Rrows, Dc, Dc, 0);

  (void)hipMemsetAsync(ksb, 0, Bc * Dc * sizeof(float), stream);
  colsum_kernel<<<dim3(2, 8, 8), 256, 0, stream>>>(kh, ksb);

  {
    GemmP p{};
    p.A = kh; p.a_rs = 1; p.a_cs = Dc; p.a_bs0 = Sc * Dc; p.a_bs1 = 64;
    p.W = vh; p.w_rs = Dc; p.w_bs0 = Sc * Dc; p.w_bs1 = 64;
    p.C = kvh; p.c_rs = 64; p.c_bs0 = Hc * 64 * 64; p.c_bs1 = 64 * 64; p.c_bf16 = 1;
    p.M = 64; p.N = 64; p.K = (int)Sc; p.act = 0; p.inner = (int)Hc;
    gemm_bf16_wmma<<<dim3(1, 1, (unsigned)(Bc * Hc)), NTHREADS, 0, stream>>>(p);
  }
  {
    int total = (int)(Bc * Kc * Hc);
    zden_kernel<<<(total + 255) / 256, 256, 0, stream>>>(sh, ksb, zb, (int)Kc, 0, total);
  }
  {
    GemmP p{};
    p.A = sh; p.a_rs = Dc; p.a_cs = 1; p.a_bs0 = 0; p.a_bs1 = 64;   // seeds shared over b
    p.W = kvh; p.w_rs = 64; p.w_bs0 = Hc * 64 * 64; p.w_bs1 = 64 * 64;
    p.rs = zb; p.rs_st = Hc; p.rs_bs0 = Kc * Hc; p.rs_bs1 = 1;
    p.C = oh; p.c_rs = Dc; p.c_bs0 = Kc * Dc; p.c_bs1 = 64; p.c_bf16 = 1;
    p.M = (int)Kc; p.N = 64; p.K = 64; p.act = 0; p.inner = (int)Hc;
    gemm_bf16_wmma<<<dim3(1, 1, (unsigned)(Bc * Hc)), NTHREADS, 0, stream>>>(p);
  }
  // out = o @ pWo + pbo  -> [B*K, D] fp32
  gemm_plain(stream, oh, Dc, pwoh, Dc, pbo, nullptr, 0,
             (float*)d_out, 0, Dc, (int)(Bc * Kc), (int)Dc, (int)Dc, 0);
}
